// _Blocks_73495480369774
// MI455X (gfx1250) — matrix operand order repaired and checked by reading, not compiled, not run on hardware
//
#include <hip/hip_runtime.h>

// MI455X / gfx1250 wave32 WMMA implementation of the 4-block gated-residual
// conv stack. All convs are lowered to v_wmma_f32_16x16x32_f16 GEMMs with f32
// accumulation; residual stream kept in WMMA C/D register layout.

typedef __attribute__((ext_vector_type(16))) _Float16 v16h;
typedef __attribute__((ext_vector_type(8)))  float    v8f;

#define Tn     4096
#define REALC  112      // real time-columns per tile
#define NCOLS  128      // 16 halo + 112 real
#define NTILES 37       // ceil(4096/112)
#define GP     72       // sG/sU pitch (halves): 144B rows, 16B aligned, spreads banks
#define WP     136      // 64x128 weight pitch (halves): 272B rows
#define OPC    72       // 64x64 weight pitch (halves)

union H16u { v16h v; float4 f[2]; };
union PK   { _Float16 h[2]; unsigned int u; };

__device__ inline v16h ld_frag(const _Float16* p0, const _Float16* p1) {
  H16u r;
  r.f[0] = *(const float4*)p0;   // ds_load_b128
  r.f[1] = *(const float4*)p1;   // ds_load_b128
  return r.v;
}

__global__ __launch_bounds__(256)
void wavenet_fused_kernel(const float* __restrict__ x,
                          const float* __restrict__ sig_w, const float* __restrict__ sig_b,
                          const float* __restrict__ gate_w, const float* __restrict__ gate_b,
                          const float* __restrict__ out_w, const float* __restrict__ out_b,
                          const float* __restrict__ skip_w, const float* __restrict__ skip_b,
                          const float* __restrict__ dil_w, const float* __restrict__ dil_b,
                          float* __restrict__ out)
{
  __shared__ __align__(16) _Float16 sG[NCOLS * GP];   // g, f16, [time][channel]
  __shared__ __align__(16) _Float16 sU[NCOLS * GP];   // gated activations
  __shared__ __align__(16) _Float16 sWa[64 * WP];     // sig_w / dil_w  [h][k=tap*64+c]
  __shared__ __align__(16) _Float16 sWb[64 * WP];     // gate_w
  __shared__ __align__(16) _Float16 sWo[64 * OPC];    // out_w (1x1)
  __shared__ float sBias[192];

  const int tid  = threadIdx.x;
  const int wv   = tid >> 5;          // wave id (0..7) -> 16-column chunk
  const int lane = tid & 31;
  const int ln   = lane & 15;         // N (column) / M (row) index within 16
  const int hi   = lane >> 4;         // lane half
  const int lc   = wv << 4;           // this wave's column base in tile
  const int tile = blockIdx.x;
  const int b    = blockIdx.y;
  const int ts   = tile * REALC;
  const int tg   = ts - 16 + lc + ln; // global t of this lane's column

  // ---- init residual stream g (f32 in C/D layout regs) + f16 LDS copy ----
  v8f gReg[4];
  #pragma unroll
  for (int m = 0; m < 4; ++m) {
    v8f g;
    #pragma unroll
    for (int r = 0; r < 8; ++r) {
      int c = m * 16 + r + hi * 8;
      float v = 0.f;
      if (tg >= 0 && tg < Tn) v = x[((size_t)b * 64 + c) * Tn + tg];
      g[r] = v;
    }
    gReg[m] = g;
    #pragma unroll
    for (int s = 0; s < 4; ++s) {
      PK pk; pk.h[0] = (_Float16)g[2 * s]; pk.h[1] = (_Float16)g[2 * s + 1];
      *(unsigned int*)&sG[(lc + ln) * GP + m * 16 + hi * 8 + 2 * s] = pk.u;
    }
  }

  const float keep = (tg >= 0) ? 1.f : 0.f;   // emulate causal zero-pad (t<0 stays 0)

  // =========================== 4 gated residual blocks ====================
  for (int blk = 0; blk < 4; ++blk) {
    __syncthreads();                       // prior sG writes & sW readers done
    {   // stage this block's weights (f32 -> f16, A-layout rows [h][k])
      const float* swp = sig_w  + blk * 8192;
      const float* gwp = gate_w + blk * 8192;
      for (int i = tid; i < 8192; i += 256) {
        int h = i >> 7, rem = i & 127, c = rem >> 1, kk = rem & 1;
        sWa[h * WP + kk * 64 + c] = (_Float16)swp[i];
        sWb[h * WP + kk * 64 + c] = (_Float16)gwp[i];
      }
      const float* owp = out_w + blk * 4096;
      for (int i = tid; i < 4096; i += 256) {
        int co = i >> 6, h = i & 63;
        sWo[co * OPC + h] = (_Float16)owp[i];
      }
      if (tid < 64) {
        sBias[tid]       = sig_b [blk * 64 + tid];
        sBias[64 + tid]  = gate_b[blk * 64 + tid];
        sBias[128 + tid] = out_b [blk * 64 + tid];
      }
    }
    __syncthreads();

    // B fragments: rows k<64 => g[c][t-1] (tap0), k>=64 => g[c][t] (tap1)
    v16h Bf[4];
    #pragma unroll
    for (int kc = 0; kc < 4; ++kc) {
      int le = lc + ln - (kc < 2 ? 1 : 0);
      if (le < 0) le = 0;                          // halo-left garbage, never valid-used
      int cb = (kc & 1) * 32 + hi * 8;
      const _Float16* p = &sG[le * GP + cb];
      Bf[kc] = ld_frag(p, p + 16);
    }
    __syncthreads();                               // all B captured before sG rewrite

    // ---- sig & gate GEMMs (shared B), gated nonlinearity -> sU -----------
    #pragma unroll
    for (int m = 0; m < 4; ++m) {
      v8f aS = {}; v8f aG = {};
      #pragma unroll
      for (int kc = 0; kc < 4; ++kc) {
        const _Float16* pa = &sWa[(m * 16 + ln) * WP + kc * 32 + hi * 8];
        v16h As = ld_frag(pa, pa + 16);
        aS = __builtin_amdgcn_wmma_f32_16x16x32_f16(false, As, false, Bf[kc],
                                                    (short)0, aS, false, false);
        const _Float16* pg = &sWb[(m * 16 + ln) * WP + kc * 32 + hi * 8];
        v16h Ag = ld_frag(pg, pg + 16);
        aG = __builtin_amdgcn_wmma_f32_16x16x32_f16(false, Ag, false, Bf[kc],
                                                    (short)0, aG, false, false);
      }
      #pragma unroll
      for (int s = 0; s < 4; ++s) {
        float u01[2];
        #pragma unroll
        for (int q = 0; q < 2; ++q) {
          int r = 2 * s + q;
          float sv = aS[r] + sBias[m * 16 + r + hi * 8];
          float gv = aG[r] + sBias[64 + m * 16 + r + hi * 8];
          sv = fmaxf(sv, 0.f);
          float sg = 1.f / (1.f + __expf(-gv));
          u01[q] = sv * sg;
        }
        PK pk; pk.h[0] = (_Float16)u01[0]; pk.h[1] = (_Float16)u01[1];
        *(unsigned int*)&sU[(lc + ln) * GP + m * 16 + hi * 8 + 2 * s] = pk.u;
      }
    }
    __syncthreads();

    // ---- 1x1 out conv, residual folded in as WMMA C operand --------------
    #pragma unroll
    for (int m = 0; m < 4; ++m) {
      v8f acc = gReg[m];
      #pragma unroll
      for (int r = 0; r < 8; ++r) acc[r] += sBias[128 + m * 16 + r + hi * 8];
      #pragma unroll
      for (int kc = 0; kc < 2; ++kc) {
        const _Float16* pa = &sWo[(m * 16 + ln) * OPC + kc * 32 + hi * 8];
        v16h Ao = ld_frag(pa, pa + 16);
        const _Float16* pu = &sU[(lc + ln) * GP + kc * 32 + hi * 8];
        v16h Bu = ld_frag(pu, pu + 16);
        acc = __builtin_amdgcn_wmma_f32_16x16x32_f16(false, Ao, false, Bu,
                                                     (short)0, acc, false, false);
      }
      #pragma unroll
      for (int r = 0; r < 8; ++r) acc[r] *= keep;  // keep t<0 columns at exactly 0
      gReg[m] = acc;
      #pragma unroll
      for (int s = 0; s < 4; ++s) {
        PK pk; pk.h[0] = (_Float16)acc[2 * s]; pk.h[1] = (_Float16)acc[2 * s + 1];
        *(unsigned int*)&sG[(lc + ln) * GP + m * 16 + hi * 8 + 2 * s] = pk.u;
      }
    }
  }

  // ============================ epilogue ==================================
  __syncthreads();
  for (int i = tid; i < 8192; i += 256) {          // stage dil_w into sWa
    int co = i >> 7, rem = i & 127, ci = rem >> 1, kk = rem & 1;
    sWa[co * WP + kk * 64 + ci] = (_Float16)dil_w[i];
  }
  if (tid < 64) sBias[tid] = dil_b[tid];
  __syncthreads();

  if (wv < 4) {
    // Dilated stride-2 conv: y[c,j] = W0 @ g[:,2j-1] + W1 @ g[:,2j] + b
    const int jl = wv * 16 + ln;                   // local output column
    v16h Bd[4];
    #pragma unroll
    for (int kc = 0; kc < 4; ++kc) {
      int le = 2 * jl + (kc < 2 ? 15 : 16);        // l for t = 2j-1 / 2j
      if (le > NCOLS - 1) le = NCOLS - 1;
      int cb = (kc & 1) * 32 + hi * 8;
      const _Float16* p = &sG[le * GP + cb];
      Bd[kc] = ld_frag(p, p + 16);
    }
    const int jg = ts / 2 + jl;
    const bool ok = (jl < REALC / 2) && (jg < Tn / 2);
    #pragma unroll
    for (int m = 0; m < 4; ++m) {
      v8f acc;
      #pragma unroll
      for (int r = 0; r < 8; ++r) acc[r] = sBias[m * 16 + r + hi * 8];
      #pragma unroll
      for (int kc = 0; kc < 4; ++kc) {
        const _Float16* pa = &sWa[(m * 16 + ln) * WP + kc * 32 + hi * 8];
        v16h Ad = ld_frag(pa, pa + 16);
        acc = __builtin_amdgcn_wmma_f32_16x16x32_f16(false, Ad, false, Bd[kc],
                                                     (short)0, acc, false, false);
      }
      if (ok) {
        #pragma unroll
        for (int r = 0; r < 8; ++r) {
          int c = m * 16 + r + hi * 8;
          out[((size_t)b * 64 + c) * (Tn / 2) + jg] = acc[r];
        }
      }
    }
  } else if (wv == 4) {
    // Skip 1x1 conv on the final timestep t = T-1 (owning tile only)
    if (ts <= Tn - 1 && Tn - 1 < ts + REALC) {
      const int ll = (Tn - 1) - (ts - 16);
      float a0 = skip_b[lane], a1 = skip_b[lane + 32];
      for (int c = 0; c < 64; ++c) {
        float gc = (float)sG[ll * GP + c];
        a0 += skip_w[lane * 64 + c] * gc;
        a1 += skip_w[(lane + 32) * 64 + c] * gc;
      }
      const size_t off = (size_t)256 * 64 * (Tn / 2);
      out[off + (size_t)b * 64 + lane]      = a0;
      out[off + (size_t)b * 64 + lane + 32] = a1;
    }
  }
}

extern "C" void kernel_launch(void* const* d_in, const int* in_sizes, int n_in,
                              void* d_out, int out_size, void* d_ws, size_t ws_size,
                              hipStream_t stream) {
  dim3 grid(NTILES, 256);
  wavenet_fused_kernel<<<grid, 256, 0, stream>>>(
      (const float*)d_in[0],
      (const float*)d_in[1], (const float*)d_in[2],
      (const float*)d_in[3], (const float*)d_in[4],
      (const float*)d_in[5], (const float*)d_in[6],
      (const float*)d_in[7], (const float*)d_in[8],
      (const float*)d_in[9], (const float*)d_in[10],
      (float*)d_out);
}
